// DotAttention_27650999451863
// MI455X (gfx1250) — matrix operand order repaired and checked by reading, not compiled, not run on hardware
//
#include <hip/hip_runtime.h>

// ---------------------------------------------------------------------------
// DotAttention for MI455X (gfx1250, wave32, WMMA bf16 16x16x32 + TDM)
//   k_cvt_w   : W fp32 -> bf16 (ws)
//   k_gemm_ln : x = inp @ W^T, fused LayerNorm -> targetT bf16 (ws)
//               (W tile fetched by Tensor Data Mover)
//   k_scores  : scores = targetT @ ctx^T -> attn region of d_out (fp32)
//               (double-buffered; A tile via TDM, B tile fp32->bf16 on VALU)
//   k_softmax : softmax over S, in place
//   k_wctx    : out = attn @ values -> [T,B,D] region (double-buffered)
// ---------------------------------------------------------------------------

typedef __attribute__((ext_vector_type(16))) __bf16 bf16x16;
typedef __attribute__((ext_vector_type(8)))  __bf16 bf16x8;
typedef __attribute__((ext_vector_type(8)))  float  f32x8;
typedef __attribute__((ext_vector_type(4)))  unsigned int u32x4;
typedef __attribute__((ext_vector_type(8)))  int    i32x8;
typedef __attribute__((ext_vector_type(4)))  int    i32x4;

#define T_LEN 1024
#define B_SZ  32
#define S_LEN 2048
#define D_DIM 512
#define LDK   40   // 32 K-elements + 8 pad (bf16) -> 80B row stride, 16B aligned

__device__ __forceinline__ unsigned pk_bf16(float a, float b) {
  unsigned ua = __float_as_uint(a), ub = __float_as_uint(b);
  ua += 0x7FFFu + ((ua >> 16) & 1u);           // round-to-nearest-even
  ub += 0x7FFFu + ((ub >> 16) & 1u);
  return (ua >> 16) | (ub & 0xFFFF0000u);
}
__device__ __forceinline__ unsigned short f2bf(float a) {
  unsigned ua = __float_as_uint(a);
  ua += 0x7FFFu + ((ua >> 16) & 1u);
  return (unsigned short)(ua >> 16);
}

union FragU { struct { bf16x8 lo, hi; } h; bf16x16 v; };

// A fragment 16x32 bf16: lane -> row = lane&15, half = lane>>4
__device__ __forceinline__ bf16x16 ld_frag_a(const unsigned short* tile, int lane) {
  const int row = lane & 15, hf = lane >> 4;
  FragU f;
  f.h.lo = *(const bf16x8*)(tile + row * LDK + hf * 8);
  f.h.hi = *(const bf16x8*)(tile + row * LDK + 16 + hf * 8);
  return f.v;
}
// B fragment 32x16 bf16 (stored [n][k] K-contiguous in LDS)
__device__ __forceinline__ bf16x16 ld_frag_b(const unsigned short* tile, int lane) {
  const int col = lane & 15, hf = lane >> 4;
  FragU f;
  f.h.lo = *(const bf16x8*)(tile + col * LDK + hf * 8);
  f.h.hi = *(const bf16x8*)(tile + col * LDK + 16 + hf * 8);
  return f.v;
}

__device__ __forceinline__ f32x8 wmma_bf16(bf16x16 a, bf16x16 b, f32x8 c) {
  return __builtin_amdgcn_wmma_f32_16x16x32_bf16(false, a, false, b, (short)0, c,
                                                 false, false);
}

// ---------------------------------------------------------------------------
// TDM: DMA a [tile_rows x 32] bf16 tile (row stride 512 elements) from global
// into LDS, padding each 16-DWORD row with 4 DWORDs -> LDK=40 element rows.
// D# group0: count=1, lds_addr, 57-bit global addr, type=2.
// D# group1: data_size=2B, pad_enable, pad_interval=16dw(code 3),
//            pad_amount=4dw(code 3), tensor_dim0=512, tile_dim0=32,
//            tile_dim1=tile_rows, stride0=512.
// ---------------------------------------------------------------------------
__device__ __forceinline__ void tdm_load_bf16_tile(const unsigned short* gptr,
                                                   unsigned lds_byte,
                                                   int tile_rows) {
  unsigned long long ga = (unsigned long long)(size_t)gptr;
  const unsigned tensor_rows = 1u << 20;   // large: tile always in-bounds
  u32x4 g0;
  g0[0] = 1u;                                               // count=1
  g0[1] = lds_byte;                                         // lds_addr
  g0[2] = (unsigned)(ga & 0xFFFFFFFFu);                     // global_addr lo
  g0[3] = (unsigned)((ga >> 32) & 0x1FFFFFFu) | (2u << 30); // addr hi | type=2
  i32x8 g1;
  g1[0] = (int)((1u << 16) | (1u << 20) | (3u << 22) | (3u << 25));
  g1[1] = (int)(512u << 16);                                // tensor_dim0 lo16
  g1[2] = (int)((tensor_rows & 0xFFFFu) << 16);             // dim0 hi | dim1 lo
  g1[3] = (int)(((tensor_rows >> 16) & 0xFFFFu) | (32u << 16)); // dim1 hi | tile_dim0
  g1[4] = (int)((unsigned)tile_rows & 0xFFFFu);             // tile_dim1 | tile_dim2=0
  g1[5] = 512;                                              // stride0 lo32
  g1[6] = 0;                                                // stride0 hi | stride1 lo
  g1[7] = 0;
  i32x4 z4 = {0, 0, 0, 0};
#if __clang_major__ >= 23
  i32x8 z8 = {0, 0, 0, 0, 0, 0, 0, 0};
  __builtin_amdgcn_tensor_load_to_lds(g0, g1, z4, z4, z8, 0);
#else
  __builtin_amdgcn_tensor_load_to_lds(g0, g1, z4, z4, 0);
#endif
}

// ------------------------------- kernel 0 ----------------------------------
__global__ __launch_bounds__(256) void k_cvt_w(const float* __restrict__ W,
                                               unsigned short* __restrict__ Wb) {
  int i = (blockIdx.x * 256 + threadIdx.x) * 4;
  float4 v = *(const float4*)(W + i);
  uint2 p; p.x = pk_bf16(v.x, v.y); p.y = pk_bf16(v.z, v.w);
  *(uint2*)(Wb + i) = p;
}

// ------------------------------- kernel 1 ----------------------------------
// x[m][e] = sum_d inp[m][d] * W[e][d], m = t*B + b ; fused LayerNorm over e,
// store bf16 targetT[(b*T + t)*D + e]. Block: 64 rows x full N=512.
__global__ __launch_bounds__(256) void k_gemm_ln(
    const float* __restrict__ inp,          // [T*B, D] fp32
    const unsigned short* __restrict__ Wb,  // [D, D] bf16
    const float* __restrict__ gamma,
    const float* __restrict__ beta,
    unsigned short* __restrict__ tT)        // [B, T, D] bf16
{
  __shared__ unsigned short At[64 * LDK];    // 5.0 KB
  __shared__ unsigned short Bt[512 * LDK];   // 40 KB (TDM destination)
  __shared__ float wsum[8][64], wsumsq[8][64];
  __shared__ float rmu[64], rrs[64];

  const int m0   = blockIdx.x * 64;
  const int tid  = threadIdx.x;
  const int lane = tid & 31;
  const int wave = tid >> 5;                 // wave owns cols [wave*64, +64)

  f32x8 acc[4][4];
  for (int mi = 0; mi < 4; ++mi)
    for (int ni = 0; ni < 4; ++ni)
      for (int v = 0; v < 8; ++v) acc[mi][ni][v] = 0.0f;

  for (int k0 = 0; k0 < D_DIM; k0 += 32) {
    // B: W bf16 512x32 tile via Tensor Data Mover (wave 0 issues)
    if (wave == 0)
      tdm_load_bf16_tile(Wb + k0, (unsigned)(size_t)&Bt[0], 512);
    // A: 64x32 fp32 -> bf16 LDS (all threads)
#pragma unroll
    for (int i = 0; i < 2; ++i) {
      int c = tid * 2 + i;
      int row = c >> 3, colc = (c & 7) * 4;
      float4 v = *(const float4*)(inp + (size_t)(m0 + row) * D_DIM + k0 + colc);
      uint2 p; p.x = pk_bf16(v.x, v.y); p.y = pk_bf16(v.z, v.w);
      *(uint2*)(&At[row * LDK + colc]) = p;
    }
    if (wave == 0) __builtin_amdgcn_s_wait_tensorcnt(0);
    __syncthreads();
    bf16x16 fa[4];
#pragma unroll
    for (int mi = 0; mi < 4; ++mi) fa[mi] = ld_frag_a(&At[(mi * 16) * LDK], lane);
#pragma unroll
    for (int ni = 0; ni < 4; ++ni) {
      bf16x16 fb = ld_frag_b(&Bt[(wave * 64 + ni * 16) * LDK], lane);
#pragma unroll
      for (int mi = 0; mi < 4; ++mi)
        acc[mi][ni] = wmma_bf16(fa[mi], fb, acc[mi][ni]);
    }
    __syncthreads();
  }

  // Deterministic LayerNorm: per-wave row partials -> fixed-order combine.
#pragma unroll
  for (int mi = 0; mi < 4; ++mi)
#pragma unroll
    for (int v = 0; v < 8; ++v) {
      float s = 0.f, s2 = 0.f;
#pragma unroll
      for (int ni = 0; ni < 4; ++ni) { float x = acc[mi][ni][v]; s += x; s2 += x * x; }
#pragma unroll
      for (int off = 1; off < 16; off <<= 1) {   // reduce within 16-lane half
        s  += __shfl_xor(s,  off, 16);
        s2 += __shfl_xor(s2, off, 16);
      }
      if ((lane & 15) == 0) {
        int row = mi * 16 + (lane >> 4) * 8 + v;
        wsum[wave][row] = s; wsumsq[wave][row] = s2;
      }
    }
  __syncthreads();
  if (tid < 64) {
    float s = 0.f, s2 = 0.f;
#pragma unroll
    for (int w = 0; w < 8; ++w) { s += wsum[w][tid]; s2 += wsumsq[w][tid]; }
    float mu  = s * (1.0f / 512.0f);
    float var = s2 * (1.0f / 512.0f) - mu * mu;
    rmu[tid] = mu;
    rrs[tid] = rsqrtf(var + 1e-6f);
  }
  __syncthreads();

  const int colL = lane & 15, hf = lane >> 4;
#pragma unroll
  for (int ni = 0; ni < 4; ++ni) {
    int col = wave * 64 + ni * 16 + colL;
    float g = gamma[col], be = beta[col];
#pragma unroll
    for (int mi = 0; mi < 4; ++mi)
#pragma unroll
      for (int v = 0; v < 8; ++v) {
        int row = mi * 16 + hf * 8 + v;
        float y = g * (acc[mi][ni][v] - rmu[row]) * rrs[row] + be;
        int m = m0 + row;
        int t = m >> 5, bb = m & 31;                 // m = t*B + b, B = 32
        tT[((size_t)bb * T_LEN + t) * D_DIM + col] = f2bf(y);
      }
  }
}

// ------------------------------- kernel 2 ----------------------------------
// attn[b*T+t][s] = sum_d targetT[b][t][d] * ctx[b][s][d].
// 128x128 tile, BK=32, double-buffered LDS; A tile via TDM.
__global__ __launch_bounds__(256) void k_scores(
    const unsigned short* __restrict__ tT,  // [B,T,D] bf16
    const float* __restrict__ ctx,          // [B,S,D] fp32
    float* __restrict__ attn)               // [B*T, S] fp32
{
  __shared__ unsigned short At[2][128 * LDK];
  __shared__ unsigned short Bt[2][128 * LDK];

  const int blk = blockIdx.x;
  const int b   = blk >> 7;                 // / (8 * 16)
  const int r   = blk & 127;
  const int m0  = (r >> 4) * 128;
  const int n0  = (r & 15) * 128;
  const int tid = threadIdx.x, lane = tid & 31, wave = tid >> 5;
  const int waveM = wave >> 1, waveN = wave & 1;  // 4 M-waves x 2 N-waves

  f32x8 acc[2][4];
  for (int mi = 0; mi < 2; ++mi)
    for (int ni = 0; ni < 4; ++ni)
      for (int v = 0; v < 8; ++v) acc[mi][ni][v] = 0.0f;

  const unsigned short* aG = tT + ((size_t)b * T_LEN + m0) * D_DIM;
  const float* bG = ctx + ((size_t)b * S_LEN + n0) * D_DIM;

  auto load_tiles = [&](int k0, int buf) {
    if (wave == 0)                                  // A: TDM, 128x32 bf16
      tdm_load_bf16_tile(aG + k0, (unsigned)(size_t)&At[buf][0], 128);
#pragma unroll
    for (int i = 0; i < 4; ++i) {                   // B: fp32 -> bf16 cvt
      int c = tid * 4 + i;
      int row = c >> 3, colc = (c & 7) * 4;
      float4 v = *(const float4*)(bG + (size_t)row * D_DIM + k0 + colc);
      uint2 p; p.x = pk_bf16(v.x, v.y); p.y = pk_bf16(v.z, v.w);
      *(uint2*)(&Bt[buf][row * LDK + colc]) = p;
    }
  };

  load_tiles(0, 0);
  for (int kk = 0; kk < D_DIM / 32; ++kk) {
    const int cur = kk & 1;
    if (wave == 0) __builtin_amdgcn_s_wait_tensorcnt(0);
    __syncthreads();                                 // tile[cur] published
    if (kk + 1 < D_DIM / 32) load_tiles((kk + 1) * 32, cur ^ 1);
    bf16x16 fa[2], fb[4];
#pragma unroll
    for (int mi = 0; mi < 2; ++mi)
      fa[mi] = ld_frag_a(&At[cur][(waveM * 32 + mi * 16) * LDK], lane);
#pragma unroll
    for (int ni = 0; ni < 4; ++ni)
      fb[ni] = ld_frag_b(&Bt[cur][(waveN * 64 + ni * 16) * LDK], lane);
#pragma unroll
    for (int mi = 0; mi < 2; ++mi)
#pragma unroll
      for (int ni = 0; ni < 4; ++ni)
        acc[mi][ni] = wmma_bf16(fa[mi], fb[ni], acc[mi][ni]);
  }

  const int colL = lane & 15, hf = lane >> 4;
#pragma unroll
  for (int mi = 0; mi < 2; ++mi)
#pragma unroll
    for (int ni = 0; ni < 4; ++ni)
#pragma unroll
      for (int v = 0; v < 8; ++v) {
        int gr  = b * T_LEN + m0 + waveM * 32 + mi * 16 + hf * 8 + v;
        int col = n0 + waveN * 64 + ni * 16 + colL;
        attn[(size_t)gr * S_LEN + col] = acc[mi][ni][v];
      }
}

// ------------------------------- kernel 3 ----------------------------------
__global__ __launch_bounds__(256) void k_softmax(float* __restrict__ attn) {
  __shared__ float sred[8];
  float* p = attn + (size_t)blockIdx.x * S_LEN;
  const int tid = threadIdx.x, lane = tid & 31, wave = tid >> 5;
  float vals[8];
  float m = -3.402823466e+38f;
#pragma unroll
  for (int i = 0; i < 8; ++i) { vals[i] = p[tid + i * 256]; m = fmaxf(m, vals[i]); }
#pragma unroll
  for (int off = 16; off > 0; off >>= 1) m = fmaxf(m, __shfl_xor(m, off, 32));
  if (lane == 0) sred[wave] = m;
  __syncthreads();
  float M = sred[0];
#pragma unroll
  for (int w = 1; w < 8; ++w) M = fmaxf(M, sred[w]);
  __syncthreads();
  float s = 0.f;
#pragma unroll
  for (int i = 0; i < 8; ++i) { vals[i] = __expf(vals[i] - M); s += vals[i]; }
#pragma unroll
  for (int off = 16; off > 0; off >>= 1) s += __shfl_xor(s, off, 32);
  if (lane == 0) sred[wave] = s;
  __syncthreads();
  float tot = 0.f;
#pragma unroll
  for (int w = 0; w < 8; ++w) tot += sred[w];
  float inv = 1.0f / tot;
#pragma unroll
  for (int i = 0; i < 8; ++i) p[tid + i * 256] = vals[i] * inv;
}

// ------------------------------- kernel 4 ----------------------------------
// out[t][b][d] = sum_s attn[b*T+t][s] * values[b][s][d]. Double-buffered;
// values transposed into LDS [d][s] so the B fragment stays K-contiguous.
__global__ __launch_bounds__(256) void k_wctx(
    const float* __restrict__ attn,     // [B*T, S] (softmaxed)
    const float* __restrict__ vals_g,   // [B,S,D]
    float* __restrict__ outp)           // [T,B,D]
{
  __shared__ unsigned short At[2][128 * LDK];
  __shared__ unsigned short Bt[2][128 * LDK];
  const int blk = blockIdx.x;
  const int b = blk >> 5;                   // / (8 * 4)
  const int r = blk & 31;
  const int m0 = (r >> 2) * 128;            // t tile
  const int n0 = (r & 3) * 128;             // d tile
  const int tid = threadIdx.x, lane = tid & 31, wave = tid >> 5;
  const int waveM = wave >> 1, waveN = wave & 1;

  f32x8 acc[2][4];
  for (int mi = 0; mi < 2; ++mi)
    for (int ni = 0; ni < 4; ++ni)
      for (int v = 0; v < 8; ++v) acc[mi][ni][v] = 0.0f;

  const float* aG = attn + ((size_t)b * T_LEN + m0) * S_LEN;
  const float* bG = vals_g + (size_t)b * S_LEN * D_DIM + n0;

  auto load_tiles = [&](int k0, int buf) {
#pragma unroll
    for (int i = 0; i < 4; ++i) {           // A: attn 128x32 fp32 -> bf16
      int c = tid * 4 + i;
      int row = c >> 3, colc = (c & 7) * 4;
      float4 v = *(const float4*)(aG + (size_t)row * S_LEN + k0 + colc);
      uint2 p; p.x = pk_bf16(v.x, v.y); p.y = pk_bf16(v.z, v.w);
      *(uint2*)(&At[buf][row * LDK + colc]) = p;
    }
#pragma unroll
    for (int i = 0; i < 4; ++i) {           // B: values 32(s)x128(d) -> [d][s]
      int c = tid * 4 + i;
      int srow = c >> 5, dcol = (c & 31) * 4;
      float4 v = *(const float4*)(bG + (size_t)(k0 + srow) * D_DIM + dcol);
      Bt[buf][(dcol + 0) * LDK + srow] = f2bf(v.x);
      Bt[buf][(dcol + 1) * LDK + srow] = f2bf(v.y);
      Bt[buf][(dcol + 2) * LDK + srow] = f2bf(v.z);
      Bt[buf][(dcol + 3) * LDK + srow] = f2bf(v.w);
    }
  };

  load_tiles(0, 0);
  for (int kk = 0; kk < S_LEN / 32; ++kk) {
    const int cur = kk & 1;
    __syncthreads();                        // tile[cur] published
    if (kk + 1 < S_LEN / 32) load_tiles((kk + 1) * 32, cur ^ 1);
    bf16x16 fa[2], fb[4];
#pragma unroll
    for (int mi = 0; mi < 2; ++mi)
      fa[mi] = ld_frag_a(&At[cur][(waveM * 32 + mi * 16) * LDK], lane);
#pragma unroll
    for (int ni = 0; ni < 4; ++ni)
      fb[ni] = ld_frag_b(&Bt[cur][(waveN * 64 + ni * 16) * LDK], lane);
#pragma unroll
    for (int mi = 0; mi < 2; ++mi)
#pragma unroll
      for (int ni = 0; ni < 4; ++ni)
        acc[mi][ni] = wmma_bf16(fa[mi], fb[ni], acc[mi][ni]);
  }

  const int colL = lane & 15, hf = lane >> 4;
#pragma unroll
  for (int mi = 0; mi < 2; ++mi)
#pragma unroll
    for (int ni = 0; ni < 4; ++ni)
#pragma unroll
      for (int v = 0; v < 8; ++v) {
        int t = m0 + waveM * 32 + mi * 16 + hf * 8 + v;
        int d = n0 + waveN * 64 + ni * 16 + colL;
        outp[((size_t)t * B_SZ + b) * D_DIM + d] = acc[mi][ni][v];
      }
}

// ------------------------------- launcher ----------------------------------
extern "C" void kernel_launch(void* const* d_in, const int* in_sizes, int n_in,
                              void* d_out, int out_size, void* d_ws, size_t ws_size,
                              hipStream_t stream) {
  (void)in_sizes; (void)n_in; (void)out_size; (void)ws_size;
  const float* inp   = (const float*)d_in[0];   // [T,B,D]
  const float* ctx   = (const float*)d_in[1];   // [B,S,D]
  const float* vals  = (const float*)d_in[2];   // [B,S,D]
  const float* W     = (const float*)d_in[3];   // [D,D]
  const float* gamma = (const float*)d_in[4];   // [D]
  const float* beta  = (const float*)d_in[5];   // [D]

  float* outp = (float*)d_out;                              // [T,B,D]
  float* attn = outp + (size_t)T_LEN * B_SZ * D_DIM;        // [B*T,S]

  unsigned short* Wb = (unsigned short*)d_ws;               // D*D bf16 (0.5 MB)
  unsigned short* tT = Wb + (size_t)D_DIM * D_DIM;          // B*T*D bf16 (32 MB)

  k_cvt_w  <<<(D_DIM * D_DIM) / (256 * 4), 256, 0, stream>>>(W, Wb);
  k_gemm_ln<<<(T_LEN * B_SZ) / 64, 256, 0, stream>>>(inp, Wb, gamma, beta, tT);
  k_scores <<<B_SZ * (T_LEN / 128) * (S_LEN / 128), 256, 0, stream>>>(tT, ctx, attn);
  k_softmax<<<B_SZ * T_LEN, 256, 0, stream>>>(attn);
  k_wctx   <<<B_SZ * (T_LEN / 128) * (D_DIM / 128), 256, 0, stream>>>(attn, vals, outp);
}
